// Discriminator_79843442033275
// MI455X (gfx1250) — matrix operand order repaired and checked by reading, not compiled, not run on hardware
//
#include <hip/hip_runtime.h>
#include <hip/hip_bf16.h>

// ---------------------------------------------------------------------------
// MI455X (gfx1250, wave32) implementation.
//
// Pipeline (all on `stream`):
//   K0a: conv2_w fp32 -> fp16 (already N-major = B^T for WMMA GEMM #1)
//   K0b: T (577x512) -> T^T fp16 padded to (512 x 608)  (B^T for GEMM #2)
//   K1 : per-sample conv1 + leaky + im2col rows (fp16) + energy-diff column
//   K2 : WMMA GEMM #1: im2col(4608x512) x conv2_w^T(512->64) -> C2 fp32
//   K3 : scatter C2 -> f fp32 (512x577) and f fp16 padded (512x608), leaky
//   K4 : WMMA GEMM #2: f16pad(512x608) x T^T(608->512) -> Mmat fp32 (512x512)
//   K5 : per-sample minibatch L1 + exp reduction + 2-layer MLP head -> out
//
// GEMM kernels compute a 16x64 strip per wave (4 accumulators). All operand
// fragments for a k-step (1 A + 4 B) are loaded into distinct live registers
// BEFORE the 4 WMMAs so the compiler can clause the loads and use partial
// s_wait_loadcnt thresholds instead of a full drain per WMMA.
// ---------------------------------------------------------------------------

typedef __attribute__((ext_vector_type(16))) _Float16 v16h;
typedef __attribute__((ext_vector_type(8)))  _Float16 v8h;
typedef __attribute__((ext_vector_type(8)))  float    v8f;

#define NB   512                 // batch
#define AF   577                 // f features
#define KP   608                 // padded K for f@T (19 * 32)
#define BD   32                  // minibatch-disc out features
#define CD   16                  // minibatch-disc kernel dim

__device__ __forceinline__ float lrelu(float x) { return x >= 0.f ? x : 0.2f * x; }

// ---------------------------------------------------------------------------
// K0a: conv2_w (64,32,4,4) fp32 -> fp16 flat (64 x 512) == B^T for GEMM #1
__global__ void cvt_w2_kernel(const float* __restrict__ w, _Float16* __restrict__ bt) {
    int i = blockIdx.x * blockDim.x + threadIdx.x;
    if (i < 64 * 512) bt[i] = (_Float16)w[i];
}

// K0b: T (577 x 512) fp32 -> T^T fp16 padded (512 x 608); pad cols are zero.
__global__ void cvt_T_kernel(const float* __restrict__ T, _Float16* __restrict__ tt) {
    int i = blockIdx.x * blockDim.x + threadIdx.x;
    if (i >= NB * KP) return;
    int n = i / KP, k = i % KP;               // n = output column of f@T
    tt[i] = (k < AF) ? (_Float16)T[k * NB + n] : (_Float16)0.f;
}

// ---------------------------------------------------------------------------
// K1: one block per sample. conv1 (1->32, 4x4, VALID, leaky) on 9x9 input,
// then im2col rows for conv2-as-GEMM, plus energy-diff column + fp16 padding.
__global__ void conv1_im2col_kernel(const float* __restrict__ readout,
                                    const float* __restrict__ energy,
                                    const float* __restrict__ conv1_w,
                                    _Float16* __restrict__ Acol,     // (4608 x 512) fp16
                                    float*    __restrict__ f32,      // (512 x 577)
                                    _Float16* __restrict__ f16pad) { // (512 x 608)
    __shared__ float s_in[81];
    __shared__ float s_w[512];
    __shared__ float s_h1[1152];              // 32 x 6 x 6
    const int n = blockIdx.x;
    const int t = threadIdx.x;                // 128 threads = 4 waves

    if (t < 81) s_in[t] = readout[n * 81 + t];
    for (int i = t; i < 512; i += 128) s_w[i] = conv1_w[i];
    __syncthreads();

    for (int idx = t; idx < 1152; idx += 128) {
        int c = idx / 36, r = idx % 36, y = r / 6, x = r % 6;
        float s = 0.f;
#pragma unroll
        for (int ky = 0; ky < 4; ++ky)
#pragma unroll
            for (int kx = 0; kx < 4; ++kx)
                s += s_in[(y + ky) * 9 + x + kx] * s_w[c * 16 + ky * 4 + kx];
        s_h1[idx] = lrelu(s);
    }
    __syncthreads();

    // im2col: row = n*9 + pos (pos = y*3+x), col k = ch*16 + ky*4 + kx
    for (int e = t; e < 4608; e += 128) {
        int pos = e >> 9, k = e & 511;
        int ch = k >> 4, ky = (k >> 2) & 3, kx = k & 3;
        int py = pos / 3, px = pos % 3;
        Acol[(n * 9 + pos) * 512 + k] = (_Float16)s_h1[ch * 36 + (py + ky) * 6 + px + kx];
    }

    if (t == 0) {
        float es = 0.f;
        for (int i = 0; i < 81; ++i) es += s_in[i];
        float ed = fabsf(es - energy[n]);
        f32[n * AF + 576]    = ed;
        f16pad[n * KP + 576] = (_Float16)ed;
    }
    if (t >= 1 && t <= 31) f16pad[n * KP + 576 + t] = (_Float16)0.f;  // zero pad 577..607
}

// ---------------------------------------------------------------------------
// K2/K4: generic fp16 WMMA GEMM. C(MxN,f32) = A(MxK,f16 row-major) x Bt^T,
// where Bt is (N x K) row-major (i.e., B transposed).
// One 16x64 strip per wave (4 x 16x16 accumulators, shared A fragment).
// Requires: K % 32 == 0, N % 64 == 0, blockDim.x % 32 == 0 (full waves).
__global__ void wmma_gemm_f16_kernel(const _Float16* __restrict__ A,
                                     const _Float16* __restrict__ Bt,
                                     float* __restrict__ C,
                                     int M, int N, int K) {
    const int lane = threadIdx.x & 31;
    const int wave = threadIdx.x >> 5;
    const int ngrp = N >> 6;                  // 64-wide column groups
    const int tile = blockIdx.x * (blockDim.x >> 5) + wave;
    if (tile >= (M >> 4) * ngrp) return;      // wave-uniform: EXEC stays all-1s
    const int mt = tile / ngrp, ng = tile % ngrp;

    const int l15 = lane & 15;
    const int hi  = lane >> 4;
    const _Float16* arow = A + (mt * 16 + l15) * (long)K;
    const _Float16* brow[4];
#pragma unroll
    for (int j = 0; j < 4; ++j)
        brow[j] = Bt + (ng * 64 + j * 16 + l15) * (long)K;

    v8f acc[4] = {};
    for (int k0 = 0; k0 < K; k0 += 32) {
        // ---- load ALL fragments for this k-step into live registers first ----
        // A frag: halfs 0..7 <- K = k0 + 8*hi .. +7 ; halfs 8..15 <- K = k0+16+8*hi .. +7
        v8h a0 = *(const v8h*)(arow + k0 + hi * 8);
        v8h a1 = *(const v8h*)(arow + k0 + 16 + hi * 8);
        // B frags (from B^T): halfs 0..7 <- K = k0 + 8*hi .. +7 ; halfs 8..15 <- K = k0+16+8*hi .. +7 at column l15
        v8h b0[4], b1[4];
#pragma unroll
        for (int j = 0; j < 4; ++j) {
            b0[j] = *(const v8h*)(brow[j] + k0 + hi * 8);
            b1[j] = *(const v8h*)(brow[j] + k0 + 16 + hi * 8);
        }
        v16h a;
#pragma unroll
        for (int i = 0; i < 8; ++i) { a[i] = a0[i]; a[i + 8] = a1[i]; }
        v16h b[4];
#pragma unroll
        for (int j = 0; j < 4; ++j)
#pragma unroll
            for (int i = 0; i < 8; ++i) { b[j][i] = b0[j][i]; b[j][i + 8] = b1[j][i]; }
        // ---- 4 WMMAs sharing the A fragment ----
#pragma unroll
        for (int j = 0; j < 4; ++j)
            acc[j] = __builtin_amdgcn_wmma_f32_16x16x32_f16(
                /*neg_a=*/false, a, /*neg_b=*/false, b[j],
                /*c_mod=*/(short)0, acc[j], /*reuse_a=*/false, /*reuse_b=*/false);
    }
    // D layout: VGPR i, lane l -> (M = i + 8*hi, N = l&15)
#pragma unroll
    for (int j = 0; j < 4; ++j) {
        float* cbase = C + (long)(mt * 16 + hi * 8) * N + ng * 64 + j * 16 + l15;
#pragma unroll
        for (int i = 0; i < 8; ++i) cbase[(long)i * N] = acc[j][i];
    }
}

// ---------------------------------------------------------------------------
// K3: scatter conv2 GEMM output into f with leaky relu.
// C2 row = n*9 + pos, col = o  ->  f[n][o*9 + pos]
__global__ void scatter_f_kernel(const float* __restrict__ C2,
                                 float* __restrict__ f32,
                                 _Float16* __restrict__ f16pad) {
    int idx = blockIdx.x * blockDim.x + threadIdx.x;
    if (idx >= NB * 576) return;
    int n = idx / 576, j = idx % 576;
    int o = j / 9, pos = j % 9;
    float v = lrelu(C2[(n * 9 + pos) * 64 + o]);
    f32[n * AF + j]    = v;
    f16pad[n * KP + j] = (_Float16)v;
}

// ---------------------------------------------------------------------------
// K5: one block per sample j (256 threads = 8 waves).
// o[j,b] = sum_i exp(-sum_c |M[i,b,c]-M[j,b,c]|), then MLP head -> sigmoid.
__global__ void mbd_final_kernel(const float* __restrict__ Mmat,   // (512 x 512)
                                 const float* __restrict__ f32,    // (512 x 577)
                                 const float* __restrict__ W1,     // (32 x 609)
                                 const float* __restrict__ b1,     // (32)
                                 const float* __restrict__ W2,     // (1 x 32)
                                 const float* __restrict__ b2,     // (1)
                                 float* __restrict__ out) {
    __shared__ float s_Mj[512];
    __shared__ float s_f[AF];
    __shared__ float s_o[BD];
    __shared__ float s_red[256];
    __shared__ float s_hid[32];
    const int n = blockIdx.x, t = threadIdx.x;

    for (int i = t; i < 512; i += 256) s_Mj[i] = Mmat[n * 512 + i];
    for (int i = t; i < AF;  i += 256) s_f[i]  = f32[n * AF + i];
    __syncthreads();

    const int b = t & 31, g = t >> 5;         // feature b, i-group g (8 groups)
    float acc = 0.f;
    for (int i = g; i < NB; i += 8) {
        const float* Mi = Mmat + i * 512 + b * CD;
        const float* Mj = s_Mj + b * CD;
        float s = 0.f;
#pragma unroll
        for (int c = 0; c < CD; ++c) s += fabsf(Mi[c] - Mj[c]);
        acc += __expf(-s);                    // includes i==j term (exp(0)=1)
    }
    s_red[t] = acc;
    __syncthreads();
    if (t < BD) {
        float s = 0.f;
#pragma unroll
        for (int gg = 0; gg < 8; ++gg) s += s_red[gg * 32 + t];
        s_o[t] = s;
    }
    __syncthreads();

    // hidden = leaky(W1 @ concat(f, o) + b1): 32 outputs, dot length 609
    float hacc = 0.f;
    for (int j = g; j < AF + BD; j += 8) {
        float xv = (j < AF) ? s_f[j] : s_o[j - AF];
        hacc += W1[b * (AF + BD) + j] * xv;
    }
    s_red[t] = hacc;                          // safe: prior s_red reads fenced above
    __syncthreads();
    if (t < 32) {
        float s = b1[t];
#pragma unroll
        for (int gg = 0; gg < 8; ++gg) s += s_red[gg * 32 + t];
        s_hid[t] = lrelu(s);
    }
    __syncthreads();
    if (t == 0) {
        float z = b2[0];
        for (int h = 0; h < 32; ++h) z += W2[h] * s_hid[h];
        out[n] = 1.f / (1.f + __expf(-z));
    }
}

// ---------------------------------------------------------------------------
extern "C" void kernel_launch(void* const* d_in, const int* in_sizes, int n_in,
                              void* d_out, int out_size, void* d_ws, size_t ws_size,
                              hipStream_t stream) {
    const float* readout = (const float*)d_in[0];   // (512,1,9,9)
    const float* energy  = (const float*)d_in[1];   // (512,)
    const float* conv1_w = (const float*)d_in[2];   // (32,1,4,4)
    const float* conv2_w = (const float*)d_in[3];   // (64,32,4,4)
    const float* T       = (const float*)d_in[4];   // (577, 512)
    const float* W1      = (const float*)d_in[5];   // (32, 609)
    const float* b1      = (const float*)d_in[6];   // (32,)
    const float* W2      = (const float*)d_in[7];   // (1, 32)
    const float* b2      = (const float*)d_in[8];   // (1,)
    float* out = (float*)d_out;                     // (512,)

    // Workspace carve-up (256-B aligned regions, ~9.4 MB total)
    char*  ws  = (char*)d_ws;
    size_t off = 0;
    auto carve = [&](size_t bytes) -> char* {
        char* p = ws + off;
        off = (off + bytes + 255) & ~(size_t)255;
        return p;
    };
    _Float16* Acol   = (_Float16*)carve((size_t)4608 * 512 * 2); // im2col fp16
    _Float16* Bt2    = (_Float16*)carve((size_t)64 * 512 * 2);   // conv2_w fp16
    float*    C2     = (float*)   carve((size_t)4608 * 64 * 4);  // conv2 GEMM out
    float*    f32    = (float*)   carve((size_t)NB * AF * 4);    // f fp32
    _Float16* f16pad = (_Float16*)carve((size_t)NB * KP * 2);    // f fp16 padded
    _Float16* Tt16   = (_Float16*)carve((size_t)NB * KP * 2);    // T^T fp16 padded
    float*    Mmat   = (float*)   carve((size_t)NB * NB * 4);    // f @ T
    (void)ws_size; (void)in_sizes; (void)n_in; (void)out_size;

    // K0: operand prep
    cvt_w2_kernel<<<(64 * 512 + 255) / 256, 256, 0, stream>>>(conv2_w, Bt2);
    cvt_T_kernel <<<(NB * KP + 255) / 256, 256, 0, stream>>>(T, Tt16);

    // K1: conv1 + im2col + energy column
    conv1_im2col_kernel<<<NB, 128, 0, stream>>>(readout, energy, conv1_w,
                                                Acol, f32, f16pad);

    // K2: conv2 as WMMA GEMM: (4608 x 512) x (512 -> 64).
    //     288 16x64 wave-strips, 4 waves per 128-thread block -> 72 blocks.
    wmma_gemm_f16_kernel<<<72, 128, 0, stream>>>(Acol, Bt2, C2, 4608, 64, 512);

    // K3: leaky + scatter into f
    scatter_f_kernel<<<(NB * 576 + 255) / 256, 256, 0, stream>>>(C2, f32, f16pad);

    // K4: f @ T as WMMA GEMM: (512 x 608) x (608 -> 512).
    //     256 16x64 wave-strips, 4 waves per 128-thread block -> 64 blocks.
    wmma_gemm_f16_kernel<<<64, 128, 0, stream>>>(f16pad, Tt16, Mmat, NB, NB, KP);

    // K5: minibatch discrimination + MLP head
    mbd_final_kernel<<<NB, 256, 0, stream>>>(Mmat, f32, W1, b1, W2, b2, out);
}
